// GitSelfAttention_13443247636849
// MI455X (gfx1250) — matrix operand order repaired and checked by reading, not compiled, not run on hardware
//
#include <hip/hip_runtime.h>
#include <hip/hip_bf16.h>
#include <stdint.h>

// ---------------------------------------------------------------------------
// GIT self-attention with relative key+query position bias, MI455X (gfx1250)
// B=4, S=1024, H=16, d=64, Hd=1024.  All GEMMs via v_wmma_f32_16x16x32_bf16.
// QKV GEMM: LDS-staged A operand via global_load_async_to_lds_b128 (ASYNCcnt),
// double-buffered B fragments in registers (software pipeline).
// ---------------------------------------------------------------------------

typedef __bf16 bf16;
typedef __attribute__((ext_vector_type(16))) __bf16 v16bf;
typedef __attribute__((ext_vector_type(8)))  __bf16 v8bf;
typedef __attribute__((ext_vector_type(8)))  float  v8f;
typedef __attribute__((ext_vector_type(4)))  int    v4i;

struct Frag {
  union { v16bf v; v8bf h[2]; };
};

__device__ __forceinline__ v8f wmma_bf16(const Frag& a, const Frag& b, v8f c) {
  // (neg_a, A, neg_b, B, c_mod, C, reuse_a, reuse_b)
  return __builtin_amdgcn_wmma_f32_16x16x32_bf16(false, a.v, false, b.v,
                                                 (short)0, c, false, false);
}

__device__ __forceinline__ float rowmax16(float v) {
  v = fmaxf(v, __shfl_xor(v, 1));
  v = fmaxf(v, __shfl_xor(v, 2));
  v = fmaxf(v, __shfl_xor(v, 4));
  v = fmaxf(v, __shfl_xor(v, 8));
  return v;
}
__device__ __forceinline__ float rowsum16(float v) {
  v += __shfl_xor(v, 1);
  v += __shfl_xor(v, 2);
  v += __shfl_xor(v, 4);
  v += __shfl_xor(v, 8);
  return v;
}

#define BATCH   4
#define S_LEN   1024
#define HEADS   16
#define HDIM    64
#define HIDDEN  1024

#if __has_builtin(__builtin_amdgcn_global_load_async_to_lds_b128)
#define HAVE_ASYNC_LDS 1
#else
#define HAVE_ASYNC_LDS 0
#endif

// 16-byte global -> LDS copy (async path when available)
__device__ __forceinline__ void async_copy16(const bf16* gsrc, bf16* ldst) {
#if HAVE_ASYNC_LDS
  __builtin_amdgcn_global_load_async_to_lds_b128(
      (__attribute__((address_space(1))) v4i*)(unsigned long long)(uintptr_t)gsrc,
      (__attribute__((address_space(3))) v4i*)(unsigned)(uintptr_t)ldst,
      0, 0);
#else
  *(v8bf*)ldst = *(const v8bf*)gsrc;
#endif
}

#if HAVE_ASYNC_LDS
#define ASYNC_WAIT_4() asm volatile("s_wait_asynccnt 0x4" ::: "memory")
#define ASYNC_WAIT_0() asm volatile("s_wait_asynccnt 0x0" ::: "memory")
#else
#define ASYNC_WAIT_4()
#define ASYNC_WAIT_0()
#endif

// ---------------------------------------------------------------------------
// fp32 -> bf16 conversion (with zero padding past srcN)
// ---------------------------------------------------------------------------
__global__ void cvt_bf16_kernel(const float* __restrict__ src,
                                bf16* __restrict__ dst, int n, int srcN) {
  int i = blockIdx.x * blockDim.x + threadIdx.x;
  if (i < n) dst[i] = (bf16)((i < srcN) ? src[i] : 0.0f);
}

// ---------------------------------------------------------------------------
// Fused QKV projection:  q = X @ Wq^T + bq  (same for k, v)
// Block: 32 m-rows x 64 n-cols (one head) with 4 waves; wave = 32x16 tile,
// three output matrices.  X slab staged in double-buffered LDS (async),
// B fragments double-buffered in registers.
// Outputs: Q,K bf16 [B,H,S,64]; V stored transposed Vt bf16 [B,H,64,S].
// ---------------------------------------------------------------------------
struct BFrags { Frag q, k, v; };

__device__ __forceinline__ void loadB(BFrags& f, const bf16* wqr,
                                      const bf16* wkr, const bf16* wvr,
                                      int kb) {
  f.q.h[0] = *(const v8bf*)(wqr + kb);
  f.q.h[1] = *(const v8bf*)(wqr + kb + 8);
  f.k.h[0] = *(const v8bf*)(wkr + kb);
  f.k.h[1] = *(const v8bf*)(wkr + kb + 8);
  f.v.h[0] = *(const v8bf*)(wvr + kb);
  f.v.h[1] = *(const v8bf*)(wvr + kb + 8);
}

__device__ __forceinline__ void loadA(Frag& a0, Frag& a1,
                                      const bf16 sl[32][32], int hl, int lm) {
  a0.h[0] = *(const v8bf*)&sl[lm][hl * 16];
  a0.h[1] = *(const v8bf*)&sl[lm][hl * 16 + 8];
  a1.h[0] = *(const v8bf*)&sl[16 + lm][hl * 16];
  a1.h[1] = *(const v8bf*)&sl[16 + lm][hl * 16 + 8];
}

__device__ __forceinline__ void mm6(const Frag& a0, const Frag& a1,
                                    const BFrags& f, v8f& cq0, v8f& cq1,
                                    v8f& ck0, v8f& ck1, v8f& cv0, v8f& cv1) {
  cq0 = wmma_bf16(a0, f.q, cq0);
  cq1 = wmma_bf16(a1, f.q, cq1);
  ck0 = wmma_bf16(a0, f.k, ck0);
  ck1 = wmma_bf16(a1, f.k, ck1);
  cv0 = wmma_bf16(a0, f.v, cv0);
  cv1 = wmma_bf16(a1, f.v, cv1);
}

__global__ __launch_bounds__(128) void qkv_kernel(
    const bf16* __restrict__ X,
    const bf16* __restrict__ Wq, const bf16* __restrict__ Wk,
    const bf16* __restrict__ Wv,
    const float* __restrict__ bq, const float* __restrict__ bk,
    const float* __restrict__ bv,
    bf16* __restrict__ Q, bf16* __restrict__ K, bf16* __restrict__ Vt) {
  __shared__ bf16 slab[2][32][32];

  const int lane = threadIdx.x & 31;
  const int w = threadIdx.x >> 5;
  const int hl = lane >> 4;
  const int lm = lane & 15;
  const int mb = blockIdx.x >> 4;        // 128 m-blocks of 32 rows
  const int nb = blockIdx.x & 15;        // 16 n-blocks of 64 cols (one head)
  const int m0 = mb * 32;
  const int n0 = nb * 64 + w * 16;

  const bf16* wqr = Wq + (size_t)(n0 + lm) * HIDDEN;
  const bf16* wkr = Wk + (size_t)(n0 + lm) * HIDDEN;
  const bf16* wvr = Wv + (size_t)(n0 + lm) * HIDDEN;

  // stage X slab [32 rows x 32 k] for one k-step into LDS buffer `buf`
  auto issueX = [&](int buf, int k0) {
#pragma unroll
    for (int j = 0; j < 4; ++j) {
      const int c = j * 32 + lane;       // 128 chunks of 16B
      const int row = c >> 2;
      const int cc = (c & 3) * 8;
      async_copy16(X + (size_t)(m0 + row) * HIDDEN + k0 + cc,
                   &slab[buf][row][cc]);
    }
  };

  BFrags fb0, fb1;
  if (w == 0) issueX(0, 0);
  loadB(fb0, wqr, wkr, wvr, hl * 16);

  const v8f zero = {};
  v8f cq0 = zero, cq1 = zero, ck0 = zero, ck1 = zero, cv0 = zero, cv1 = zero;

  for (int k0 = 0; k0 < HIDDEN; k0 += 64) {
    // ---- substep A: consume slab[0]/fb0, prefetch slab[1]/fb1 @ k0+32
    __syncthreads();                       // readers of slab[1] are done
    if (w == 0) {
      issueX(1, k0 + 32);                  // k0+32 <= 992 always valid
      ASYNC_WAIT_4();                      // slab[0] batch complete
    }
    __syncthreads();                       // publish slab[0]
    loadB(fb1, wqr, wkr, wvr, k0 + 32 + hl * 16);
    {
      Frag a0, a1;
      loadA(a0, a1, slab[0], hl, lm);
      mm6(a0, a1, fb0, cq0, cq1, ck0, ck1, cv0, cv1);
    }

    // ---- substep B: consume slab[1]/fb1, prefetch slab[0]/fb0 @ k0+64
    const bool more = (k0 + 64) < HIDDEN;
    __syncthreads();                       // readers of slab[0] are done
    if (w == 0) {
      if (more) {
        issueX(0, k0 + 64);
        ASYNC_WAIT_4();                    // slab[1] batch complete
      } else {
        ASYNC_WAIT_0();
      }
    }
    __syncthreads();                       // publish slab[1]
    if (more) loadB(fb0, wqr, wkr, wvr, k0 + 64 + hl * 16);
    {
      Frag a0, a1;
      loadA(a0, a1, slab[1], hl, lm);
      mm6(a0, a1, fb1, cq0, cq1, ck0, ck1, cv0, cv1);
    }
  }

  const float biq = bq[n0 + lm], bik = bk[n0 + lm], biv = bv[n0 + lm];
  const int h = n0 >> 6;
  const int d = (n0 & 63) + lm;
  const int b = m0 >> 10;
#pragma unroll
  for (int ms = 0; ms < 2; ++ms) {
#pragma unroll
    for (int p = 0; p < 8; ++p) {
      const int s = (m0 & 1023) + ms * 16 + p + hl * 8;
      const size_t qi = (((size_t)b * HEADS + h) * S_LEN + s) * HDIM + d;
      const float q_ = (ms ? cq1[p] : cq0[p]) + biq;
      const float k_ = (ms ? ck1[p] : ck0[p]) + bik;
      const float v_ = (ms ? cv1[p] : cv0[p]) + biv;
      Q[qi] = (bf16)q_;
      K[qi] = (bf16)k_;
      const size_t vi = (((size_t)b * HEADS + h) * HDIM + d) * S_LEN + s;
      Vt[vi] = (bf16)v_;
    }
  }
}

// ---------------------------------------------------------------------------
// Flash attention with relative key+query bias.
// One wave handles one (b,h, 16-row l-tile); loops over r in steps of 32.
// bias[i,j] = G[i][t] + H[j][t], t = i-j+15, computed via WMMA against the
// 32-row dist_emb band starting at base = l0 - r1 + 1008 (Dist padded to 2048).
// ---------------------------------------------------------------------------
__global__ __launch_bounds__(128) void attn_kernel(
    const bf16* __restrict__ Q, const bf16* __restrict__ K,
    const bf16* __restrict__ Vt, const bf16* __restrict__ Dist,  // [2048,64]
    float* __restrict__ Out) {                                   // [B,S,H,64]
  __shared__ float Gs[4][16][33];
  __shared__ float Hs[4][16][33];
  __shared__ bf16  Ps[4][16][32];

  const int lane = threadIdx.x & 31;
  const int w = threadIdx.x >> 5;
  const int hl = lane >> 4;
  const int lm = lane & 15;
  const int unit = blockIdx.x * 4 + w;   // 4096 wave units
  const int lt = unit & 63;
  const int bh = unit >> 6;              // b*16 + h
  const int l0 = lt * 16;

  const bf16* Qb = Q  + (size_t)bh * S_LEN * HDIM;
  const bf16* Kb = K  + (size_t)bh * S_LEN * HDIM;
  const bf16* Vb = Vt + (size_t)bh * HDIM * S_LEN;

  // Q tile as two A-fragments (K = d dimension, 64 -> 2 k-steps)
  Frag aq0, aq1;
  {
    const bf16* qr = Qb + (size_t)(l0 + lm) * HDIM;
    aq0.h[0] = *(const v8bf*)(qr + hl * 16);
    aq0.h[1] = *(const v8bf*)(qr + hl * 16 + 8);
    aq1.h[0] = *(const v8bf*)(qr + 32 + hl * 16);
    aq1.h[1] = *(const v8bf*)(qr + 32 + hl * 16 + 8);
  }

  const v8f zero = {};
  v8f acc[4] = {zero, zero, zero, zero};   // ctx accumulators, 4 d-subtiles
  float mrow[8], lrow[8];
#pragma unroll
  for (int p = 0; p < 8; ++p) { mrow[p] = -1.0e30f; lrow[p] = 0.0f; }

  for (int r0 = 0; r0 < S_LEN; r0 += 32) {
    v8f sfull[2];
#pragma unroll
    for (int u = 0; u < 2; ++u) {
      const int r1 = r0 + u * 16;
      // ---- K tile as B-fragments (Q@K^T) and A-fragments (H = K@band^T)
      const bf16* kr = Kb + (size_t)(r1 + lm) * HDIM;
      Frag bk0, bk1, ak0, ak1;
      bk0.h[0] = *(const v8bf*)(kr + hl * 16);
      bk0.h[1] = *(const v8bf*)(kr + hl * 16 + 8);
      bk1.h[0] = *(const v8bf*)(kr + 32 + hl * 16);
      bk1.h[1] = *(const v8bf*)(kr + 32 + hl * 16 + 8);
      ak0.h[0] = *(const v8bf*)(kr + hl * 16);
      ak0.h[1] = *(const v8bf*)(kr + hl * 16 + 8);
      ak1.h[0] = *(const v8bf*)(kr + 32 + hl * 16);
      ak1.h[1] = *(const v8bf*)(kr + 32 + hl * 16 + 8);

      // ---- raw scores
      v8f s = zero;
      s = wmma_bf16(aq0, bk0, s);
      s = wmma_bf16(aq1, bk1, s);

      // ---- dist_emb band fragments (2 n-tiles x 2 k-steps)
      const int base = l0 - r1 + 1008;   // in [0, 2016]
      Frag bb[2][2];
#pragma unroll
      for (int nt = 0; nt < 2; ++nt) {
        const bf16* dr = Dist + (size_t)(base + nt * 16 + lm) * HDIM;
        bb[nt][0].h[0] = *(const v8bf*)(dr + hl * 16);
        bb[nt][0].h[1] = *(const v8bf*)(dr + hl * 16 + 8);
        bb[nt][1].h[0] = *(const v8bf*)(dr + 32 + hl * 16);
        bb[nt][1].h[1] = *(const v8bf*)(dr + 32 + hl * 16 + 8);
      }
      v8f G0 = zero, G1 = zero, H0 = zero, H1 = zero;
      G0 = wmma_bf16(aq0, bb[0][0], G0); G0 = wmma_bf16(aq1, bb[0][1], G0);
      G1 = wmma_bf16(aq0, bb[1][0], G1); G1 = wmma_bf16(aq1, bb[1][1], G1);
      H0 = wmma_bf16(ak0, bb[0][0], H0); H0 = wmma_bf16(ak1, bb[0][1], H0);
      H1 = wmma_bf16(ak0, bb[1][0], H1); H1 = wmma_bf16(ak1, bb[1][1], H1);

      // ---- spill G/H to per-wave LDS, then gather diagonal band entries
#pragma unroll
      for (int p = 0; p < 8; ++p) {
        const int row = p + hl * 8;
        Gs[w][row][lm]      = G0[p];
        Gs[w][row][lm + 16] = G1[p];
        Hs[w][row][lm]      = H0[p];
        Hs[w][row][lm + 16] = H1[p];
      }
      asm volatile("s_wait_dscnt 0x0" ::: "memory");
#pragma unroll
      for (int p = 0; p < 8; ++p) {
        const int i = p + hl * 8;
        const int t = i - lm + 15;       // 0..30
        float sv = s[p] + Gs[w][i][t] + Hs[w][lm][t];
        sfull[u][p] = sv * 0.125f;       // 1/sqrt(64)
      }
    }

    // ---- online softmax update over the 32 new columns
    float alpha[8];
#pragma unroll
    for (int p = 0; p < 8; ++p) {
      float v = fmaxf(sfull[0][p], sfull[1][p]);
      v = rowmax16(v);
      const float mn = fmaxf(mrow[p], v);
      alpha[p] = __expf(mrow[p] - mn);
      mrow[p] = mn;
    }
#pragma unroll
    for (int p = 0; p < 8; ++p) {
      const int i = p + hl * 8;
      const float p0 = __expf(sfull[0][p] - mrow[p]);
      const float p1 = __expf(sfull[1][p] - mrow[p]);
      Ps[w][i][lm]      = (bf16)p0;
      Ps[w][i][lm + 16] = (bf16)p1;
      lrow[p] = lrow[p] * alpha[p] + rowsum16(p0 + p1);
    }
#pragma unroll
    for (int dt = 0; dt < 4; ++dt)
#pragma unroll
      for (int p = 0; p < 8; ++p) acc[dt][p] *= alpha[p];

    asm volatile("s_wait_dscnt 0x0" ::: "memory");
    // ---- P tile as A-fragment (K = 32 r-columns)
    Frag ap;
    ap.h[0] = *(const v8bf*)&Ps[w][lm][hl * 16];
    ap.h[1] = *(const v8bf*)&Ps[w][lm][hl * 16 + 8];
    // ---- V^T B-fragments per d-subtile, accumulate ctx
#pragma unroll
    for (int dt = 0; dt < 4; ++dt) {
      Frag bv;
      const bf16* vr = Vb + (size_t)(dt * 16 + lm) * S_LEN + r0 + hl * 16;
      bv.h[0] = *(const v8bf*)(vr);
      bv.h[1] = *(const v8bf*)(vr + 8);
      acc[dt] = wmma_bf16(ap, bv, acc[dt]);
    }
  }

  // ---- normalize and store: out[b, s, h, d]
  const int b = bh >> 4, h = bh & 15;
#pragma unroll
  for (int dt = 0; dt < 4; ++dt) {
#pragma unroll
    for (int p = 0; p < 8; ++p) {
      const int i = p + hl * 8;
      const float val = acc[dt][p] / lrow[p];
      const size_t oi =
          (((size_t)b * S_LEN + (l0 + i)) * HEADS + h) * HDIM + dt * 16 + lm;
      Out[oi] = val;
    }
  }
}

// ---------------------------------------------------------------------------
// Launch
// ---------------------------------------------------------------------------
extern "C" void kernel_launch(void* const* d_in, const int* in_sizes, int n_in,
                              void* d_out, int out_size, void* d_ws,
                              size_t ws_size, hipStream_t stream) {
  const float* hs   = (const float*)d_in[0];
  const float* wq   = (const float*)d_in[1];
  const float* bq   = (const float*)d_in[2];
  const float* wk   = (const float*)d_in[3];
  const float* bk   = (const float*)d_in[4];
  const float* wv   = (const float*)d_in[5];
  const float* bv   = (const float*)d_in[6];
  const float* dist = (const float*)d_in[7];

  const size_t nX = (size_t)BATCH * S_LEN * HIDDEN;       // 4,194,304
  const size_t nW = (size_t)HIDDEN * HIDDEN;              // 1,048,576
  const size_t nD = (size_t)2048 * HDIM;                  // padded dist rows
  const size_t nQ = (size_t)BATCH * HEADS * S_LEN * HDIM; // 4,194,304

  bf16* X   = (bf16*)d_ws;
  bf16* Wqb = X + nX;
  bf16* Wkb = Wqb + nW;
  bf16* Wvb = Wkb + nW;
  bf16* Db  = Wvb + nW;
  bf16* Qb  = Db + nD;
  bf16* Kb  = Qb + nQ;
  bf16* Vtb = Kb + nQ;

  cvt_bf16_kernel<<<(int)((nX + 255) / 256), 256, 0, stream>>>(hs, X, (int)nX, (int)nX);
  cvt_bf16_kernel<<<(int)((nW + 255) / 256), 256, 0, stream>>>(wq, Wqb, (int)nW, (int)nW);
  cvt_bf16_kernel<<<(int)((nW + 255) / 256), 256, 0, stream>>>(wk, Wkb, (int)nW, (int)nW);
  cvt_bf16_kernel<<<(int)((nW + 255) / 256), 256, 0, stream>>>(wv, Wvb, (int)nW, (int)nW);
  cvt_bf16_kernel<<<(int)((nD + 255) / 256), 256, 0, stream>>>(dist, Db, (int)nD,
                                                               2047 * HDIM);

  // 128 m-blocks x 16 n-blocks, 4 waves per block (one 32x16 tile each)
  qkv_kernel<<<2048, 128, 0, stream>>>(X, Wqb, Wkb, Wvb, bq, bk, bv, Qb, Kb, Vtb);

  // B*H*(S/16) = 4096 wave units / 4 waves per block
  attn_kernel<<<1024, 128, 0, stream>>>(Qb, Kb, Vtb, Db, (float*)d_out);
}
